// MultimodalNaive_47004122087744
// MI455X (gfx1250) — matrix operand order repaired and checked by reading, not compiled, not run on hardware
//
#include <hip/hip_runtime.h>
#include <hip/hip_bf16.h>

// ---------------------------------------------------------------------------
// Types for CDNA5 WMMA (gfx1250, wave32)
// ---------------------------------------------------------------------------
typedef __attribute__((ext_vector_type(16))) __bf16 v16bf;
typedef __attribute__((ext_vector_type(8)))  __bf16 v8bf;
typedef __attribute__((ext_vector_type(8)))  float  v8f;
typedef __attribute__((ext_vector_type(4))) unsigned int u32x4;
typedef __attribute__((ext_vector_type(8))) int         i32x8;
typedef __attribute__((ext_vector_type(4))) int         i32x4;

#define TM 4   // 16-row (out-channel) tiles per block
#define TN 2   // 16-col (pixel) tiles per block  -> 8 waves, 256 threads

#if __has_builtin(__builtin_amdgcn_tensor_load_to_lds) && __has_builtin(__builtin_amdgcn_s_wait_tensorcnt)
#define HAVE_TDM 1
#else
#define HAVE_TDM 0
#endif

#if HAVE_TDM
// Issue a TDM load of a 2D bf16 tile [rows x 64] (row stride Kpad elements)
// from global `gsrc` into LDS at byte offset `lds_off`. Tile is 64x64 elements;
// rows beyond `rows` and cols beyond Kpad read as zero per D# OOB rules.
__device__ __forceinline__ void tdm_load_w_tile(unsigned lds_off, const __bf16* gsrc,
                                                unsigned Kpad, unsigned rows) {
  unsigned long long ga = (unsigned long long)(size_t)gsrc;
  u32x4 g0;
  g0[0] = 1u;                                            // count=1 (valid), user mode
  g0[1] = lds_off;                                       // lds_addr
  g0[2] = (unsigned)(ga & 0xffffffffu);                  // global_addr[31:0]
  g0[3] = (unsigned)((ga >> 32) & 0x01ffffffu) | (2u << 30); // global_addr[56:32], type=2
  i32x8 g1;
  g1[0] = (1 << 16);                                     // workgroup_mask=0, data_size=1 (2B)
  g1[1] = (int)((Kpad & 0xffffu) << 16);                 // tensor_dim0[15:0]
  g1[2] = (int)((Kpad >> 16) & 0xffffu) | (int)((rows & 0xffffu) << 16); // dim0 hi | dim1 lo
  g1[3] = (int)((rows >> 16) & 0xffffu) | (64 << 16);    // dim1 hi | tile_dim0=64
  g1[4] = 64;                                            // tile_dim1=64, tile_dim2=0
  g1[5] = (int)Kpad;                                     // tensor_dim0_stride[31:0]
  g1[6] = 0;                                             // stride0 hi | stride1 lo
  g1[7] = 0;
  i32x4 z4 = {0, 0, 0, 0};
#if defined(__clang_major__) && (__clang_major__ >= 23)
  i32x8 z8 = {0, 0, 0, 0, 0, 0, 0, 0};
  __builtin_amdgcn_tensor_load_to_lds(g0, g1, z4, z4, z8, 0);
#else
  __builtin_amdgcn_tensor_load_to_lds(g0, g1, z4, z4, 0);
#endif
}
#endif

// ---------------------------------------------------------------------------
// Implicit-GEMM convolution via V_WMMA_F32_16X16X32_BF16, K-step 64 (2 WMMA).
//   X  : bf16 activations [N, Cin, H, W]
//   Wt : bf16 weights, zero-padded K: [Cout, Kpad], Kpad = roundup(Cin*KH*KW, 64)
//   A (weights) tile staged to LDS by the Tensor Data Mover; B (im2col) tile
//   gathered by one wave per N-tile into LDS; epilogue fuses affine/bias,
//   residual add, ReLU; writes bf16 and/or f32.
// ---------------------------------------------------------------------------
__global__ __launch_bounds__(256) void conv_wmma_kernel(
    const __bf16* __restrict__ X, const __bf16* __restrict__ Wt,
    const float* __restrict__ scale, const float* __restrict__ bias,
    const __bf16* __restrict__ resid,
    __bf16* __restrict__ outb, float* __restrict__ outf,
    int N, int Cin, int H, int W, int Cout, int KH, int KW,
    int stride, int pad, int Ho, int Wo, int relu)
{
  __shared__ __align__(128) __bf16 ldsA[64 * 64];        // 8 KB: 64 oc x 64 K weight tile
  __shared__ __align__(128) __bf16 ldsB[TN][2][32 * 16]; // 4 KB: per-N-tile, 2 K-substeps

  const int lane = threadIdx.x & 31;
  const int wave = threadIdx.x >> 5;
  const int mw   = wave & (TM - 1);   // 0..3  (which 16-row tile)
  const int nw   = wave >> 2;         // 0..1  (which 16-col tile)

  const int Ktot = Cin * KH * KW;
  const int Kpad = (Ktot + 63) & ~63;
  const int KHW  = KH * KW;
  const bool k1x1 = (KH == 1) && (KW == 1);

  const int Npix    = N * Ho * Wo;
  const int pixBase = (blockIdx.x * TN + nw) * 16;
  const int ocBase0 = blockIdx.y * (TM * 16);
  const int ocBase  = ocBase0 + mw * 16;

  const int col  = lane & 15;   // A row / B col / D col
  const int half = lane >> 4;   // K-half selector (ISA 7.12.2 layouts)

  // Decode this lane's output pixel (shared by B gather and epilogue)
  int img = 0, oy = 0, ox = 0;
  const int pix = pixBase + col;
  const bool pixOK = (pix < Npix);
  if (pixOK) { img = pix / (Ho * Wo); int r = pix - img * (Ho * Wo); oy = r / Wo; ox = r - oy * Wo; }
  const int iy0 = oy * stride - pad;
  const int ix0 = ox * stride - pad;

#if !HAVE_TDM
  const int ocA = ocBase + col;
  const int ocl = (ocA < Cout) ? ocA : (Cout - 1);
  const __bf16* wrow = Wt + (size_t)ocl * Kpad;
#endif

  v8f acc = {};

  for (int k0 = 0; k0 < Kpad; k0 += 64) {
    // ---- stage B tile (64K x 16N) into LDS, one wave per N tile ----
    if (mw == 0) {
      #pragma unroll
      for (int s = 0; s < 2; ++s) {
        union { v16bf v; __bf16 e[16]; } bu;
        #pragma unroll
        for (int e2 = 0; e2 < 16; ++e2) {
          const int k = k0 + s * 32 + half * 8 + (e2 < 8 ? e2 : 8 + e2);  // B: table order, k = 8h+e (e<8), 16+8h+(e-8) (e>=8)
          float v = 0.f;
          if (pixOK && k < Ktot) {
            int ci, kh, kw2;
            if (k1x1) { ci = k; kh = 0; kw2 = 0; }
            else { ci = k / KHW; int rr = k - ci * KHW; kh = rr / KW; kw2 = rr - kh * KW; }
            const int iy = iy0 + kh, ix = ix0 + kw2;
            if (iy >= 0 && iy < H && ix >= 0 && ix < W)
              v = (float)X[(((size_t)img * Cin + ci) * H + iy) * W + ix];
          }
          bu.e[e2] = (__bf16)v;
        }
        *(v16bf*)&ldsB[nw][s][lane * 16] = bu.v;
      }
    }

#if HAVE_TDM
    // ---- TDM: DMA the 64x64 bf16 weight tile into LDS (one wave issues) ----
    if (wave == 0) {
      tdm_load_w_tile((unsigned)(size_t)&ldsA[0],
                      Wt + (size_t)ocBase0 * Kpad + k0,
                      (unsigned)Kpad, (unsigned)(Cout - ocBase0));
      __builtin_amdgcn_s_wait_tensorcnt(0);
    }
#endif
    __syncthreads();

    // ---- two WMMAs (K substeps of 32) ----
#if HAVE_TDM
    const int arow = mw * 16 + col;
    #pragma unroll
    for (int s = 0; s < 2; ++s) {
      union { v16bf v; v8bf h[2]; } au;
      au.h[0] = *(const v8bf*)&ldsA[arow * 64 + s * 32 + half * 8];
      au.h[1] = *(const v8bf*)&ldsA[arow * 64 + s * 32 + 16 + half * 8];
      v16bf bfrag = *(const v16bf*)&ldsB[nw][s][lane * 16];
      acc = __builtin_amdgcn_wmma_f32_16x16x32_bf16(false, au.v, false, bfrag,
                                                    (short)0, acc, false, false);
    }
#else
    #pragma unroll
    for (int s = 0; s < 2; ++s) {
      union { v16bf v; v8bf h[2]; } au;
      au.h[0] = *(const v8bf*)(wrow + k0 + s * 32 + half * 8);
      au.h[1] = *(const v8bf*)(wrow + k0 + s * 32 + 16 + half * 8);
      v16bf bfrag = *(const v16bf*)&ldsB[nw][s][lane * 16];
      acc = __builtin_amdgcn_wmma_f32_16x16x32_bf16(false, au.v, false, bfrag,
                                                    (short)0, acc, false, false);
    }
    if (k0 + 64 < Kpad) __builtin_prefetch(wrow + k0 + 64, 0, 1);
#endif
    __syncthreads();
  }

  // ---- fused epilogue: affine / bias, residual, relu, store ----
  if (pixOK) {
    const int mBase = half * 8;   // D layout: lanes 16-31 hold M = r + 8
    #pragma unroll
    for (int r = 0; r < 8; ++r) {
      const int o = ocBase + mBase + r;
      if (o < Cout) {
        float v = acc[r];
        if (scale)      v = v * scale[o] + bias[o];
        else if (bias)  v += bias[o];
        const size_t oidx = (((size_t)img * Cout + o) * Ho + oy) * Wo + ox;
        if (resid) v += (float)resid[oidx];
        if (relu)  v = fmaxf(v, 0.f);
        if (outb)  outb[oidx] = (__bf16)v;
        if (outf)  outf[oidx] = v;
      }
    }
  }
}

// ---------------------------------------------------------------------------
// Helper kernels
// ---------------------------------------------------------------------------
__global__ void wconvert_kernel(const float* __restrict__ src, __bf16* __restrict__ dst,
                                int Cout, int Ktot, int Kpad) {
  int i = blockIdx.x * blockDim.x + threadIdx.x;
  int total = Cout * Kpad;
  if (i >= total) return;
  int k = i % Kpad; int o = i / Kpad;
  float v = (k < Ktot) ? src[(size_t)o * Ktot + k] : 0.f;
  dst[i] = (__bf16)v;
}

__global__ void split6_kernel(const float* __restrict__ x, __bf16* __restrict__ rgb,
                              __bf16* __restrict__ dep, int N, int HW) {
  int i = blockIdx.x * blockDim.x + threadIdx.x;
  int total = N * 3 * HW;
  if (i >= total) return;
  int hw = i % HW; int t = i / HW; int c = t % 3; int n = t / 3;
  rgb[i] = (__bf16)x[(((size_t)n * 6) + c) * HW + hw];
  dep[i] = (__bf16)x[(((size_t)n * 6) + c + 3) * HW + hw];
}

__global__ void maxpool3s2_kernel(const __bf16* __restrict__ x, __bf16* __restrict__ y,
                                  int N, int C, int H, int W, int Ho, int Wo) {
  int i = blockIdx.x * blockDim.x + threadIdx.x;
  int total = N * C * Ho * Wo;
  if (i >= total) return;
  int wo = i % Wo; int t = i / Wo; int ho = t % Ho; t /= Ho; int c = t % C; int n = t / C;
  float m = -1e30f;
  int y0 = ho * 2 - 1, x0 = wo * 2 - 1;
  for (int dy = 0; dy < 3; ++dy) {
    int iy = y0 + dy; if (iy < 0 || iy >= H) continue;
    for (int dx = 0; dx < 3; ++dx) {
      int ix = x0 + dx; if (ix < 0 || ix >= W) continue;
      float v = (float)x[(((size_t)n * C + c) * H + iy) * W + ix];
      m = fmaxf(m, v);
    }
  }
  y[i] = (__bf16)m;
}

// dst[N,C,H,W] += nearest-2x-upsample(src[N,C,H/2,W/2])
__global__ void upadd_kernel(__bf16* __restrict__ dst, const __bf16* __restrict__ src,
                             int N, int C, int H, int W) {
  int i = blockIdx.x * blockDim.x + threadIdx.x;
  int total = N * C * H * W;
  if (i >= total) return;
  int x = i % W; int t = i / W; int y = t % H; t /= H; int c = t % C; int n = t / C;
  float v = (float)dst[i] +
            (float)src[(((size_t)n * C + c) * (H / 2) + (y / 2)) * (W / 2) + (x / 2)];
  dst[i] = (__bf16)v;
}

// channel concat: dst[N,2C,HW] = cat(a[N,C,HW], b[N,C,HW])
__global__ void catcopy_kernel(const __bf16* __restrict__ a, const __bf16* __restrict__ b,
                               __bf16* __restrict__ dst, int N, int C, int HW) {
  int i = blockIdx.x * blockDim.x + threadIdx.x;
  int total = N * 2 * C * HW;
  if (i >= total) return;
  int hw = i % HW; int t = i / HW; int c = t % (2 * C); int n = t / (2 * C);
  dst[i] = (c < C) ? a[((size_t)n * C + c) * HW + hw]
                   : b[((size_t)n * C + (c - C)) * HW + hw];
}

// dst[N,C,Ho,Wo] = src[N,C,2y,2x]
__global__ void sub2_kernel(const float* __restrict__ src, float* __restrict__ dst,
                            int N, int C, int H, int W, int Ho, int Wo) {
  int i = blockIdx.x * blockDim.x + threadIdx.x;
  int total = N * C * Ho * Wo;
  if (i >= total) return;
  int x = i % Wo; int t = i / Wo; int y = t % Ho; t /= Ho; int c = t % C; int n = t / C;
  dst[i] = src[(((size_t)n * C + c) * H + 2 * y) * W + 2 * x];
}

// ---------------------------------------------------------------------------
// Host-side orchestration
// ---------------------------------------------------------------------------
struct BlkIdx { int b1,b2,b3,bd,s1,s2,s3,sd,w1,w2,w3,wd; bool down; };
struct BBIdx  { int lb[4],lw[4],ob[4],ow[4]; int stem_b,stem_s,stem_w; BlkIdx blk[16]; };

// Replicates jax tree_flatten order of the backbone pytree:
// dict keys sorted: 'fpn' < 'layers' < 'stem'; fpn keys: lb,lw,ob,ow;
// block keys sorted: b1,b2,b3,(bd),s1,s2,s3,(sd),w1,w2,w3,(wd); stem keys: b,s,w.
static void fill_bb(BBIdx& B, int base) {
  int i = base;
  for (int j = 0; j < 4; ++j) B.lb[j] = i++;
  for (int j = 0; j < 4; ++j) B.lw[j] = i++;
  for (int j = 0; j < 4; ++j) B.ob[j] = i++;
  for (int j = 0; j < 4; ++j) B.ow[j] = i++;
  const int nb[4] = {3, 4, 6, 3};
  int bi = 0;
  for (int s = 0; s < 4; ++s)
    for (int b = 0; b < nb[s]; ++b) {
      BlkIdx& K = B.blk[bi++]; K.down = (b == 0);
      K.b1 = i++; K.b2 = i++; K.b3 = i++; if (K.down) K.bd = i++;
      K.s1 = i++; K.s2 = i++; K.s3 = i++; if (K.down) K.sd = i++;
      K.w1 = i++; K.w2 = i++; K.w3 = i++; if (K.down) K.wd = i++;
    }
  B.stem_b = i++; B.stem_s = i++; B.stem_w = i++;
}

static void run_conv(hipStream_t st, const __bf16* X, int N, int Cin, int H, int W,
                     const float* Wf, int Cout, int KH, int KW, int stride, int pad,
                     const float* scale, const float* bias, const __bf16* resid,
                     __bf16* outb, float* outf, int relu, __bf16* wscr,
                     int& Ho, int& Wo) {
  int Ktot = Cin * KH * KW;
  int Kpad = (Ktot + 63) & ~63;
  int total = Cout * Kpad;
  wconvert_kernel<<<(total + 255) / 256, 256, 0, st>>>(Wf, wscr, Cout, Ktot, Kpad);
  Ho = (H + 2 * pad - KH) / stride + 1;
  Wo = (W + 2 * pad - KW) / stride + 1;
  int Npix = N * Ho * Wo;
  dim3 grid((Npix + 16 * TN - 1) / (16 * TN), (Cout + 16 * TM - 1) / (16 * TM));
  conv_wmma_kernel<<<grid, 256, 0, st>>>(X, wscr, scale, bias, resid, outb, outf,
                                         N, Cin, H, W, Cout, KH, KW, stride, pad, Ho, Wo, relu);
}

struct WSBuf { __bf16 *wscr, *x, *y, *t1, *t2, *r, *cat; };

static void run_backbone(hipStream_t st, const BBIdx& B, void* const* din,
                         const __bf16* input3, WSBuf& ws,
                         __bf16* cs[4], __bf16* lat[4], __bf16* feat[4]) {
  auto F = [&](int idx) { return (const float*)din[idx]; };
  int Ho, Wo;
  const int nb[4]   = {3, 4, 6, 3};
  const int mids[4] = {64, 128, 256, 512};
  const int ocs[4]  = {256, 512, 1024, 2048};
  const int st4[4]  = {1, 2, 2, 2};

  // stem: 7x7 s2 p3 -> [2,64,112,112], affine+relu
  run_conv(st, input3, 2, 3, 224, 224, F(B.stem_w), 64, 7, 7, 2, 3,
           F(B.stem_s), F(B.stem_b), nullptr, ws.y, nullptr, 1, ws.wscr, Ho, Wo);
  { int total = 2 * 64 * 56 * 56;
    maxpool3s2_kernel<<<(total + 255) / 256, 256, 0, st>>>(ws.y, ws.x, 2, 64, 112, 112, 56, 56); }

  __bf16* x = ws.x; __bf16* y = ws.y;
  int Cin = 64, H = 56, W = 56;
  int bi = 0;
  for (int s = 0; s < 4; ++s) {
    for (int b = 0; b < nb[s]; ++b) {
      const BlkIdx& K = B.blk[bi++];
      const int str = (b == 0) ? st4[s] : 1;
      const int mid = mids[s], oc = ocs[s];
      int h1, w1, h2, w2, hd, wd_;
      run_conv(st, x, 2, Cin, H, W, F(K.w1), mid, 1, 1, 1, 0,
               F(K.s1), F(K.b1), nullptr, ws.t1, nullptr, 1, ws.wscr, h1, w1);
      run_conv(st, ws.t1, 2, mid, h1, w1, F(K.w2), mid, 3, 3, str, 1,
               F(K.s2), F(K.b2), nullptr, ws.t2, nullptr, 1, ws.wscr, h2, w2);
      const __bf16* resid;
      if (K.down) {
        run_conv(st, x, 2, Cin, H, W, F(K.wd), oc, 1, 1, str, 0,
                 F(K.sd), F(K.bd), nullptr, ws.r, nullptr, 0, ws.wscr, hd, wd_);
        resid = ws.r;
      } else {
        resid = x;
      }
      run_conv(st, ws.t2, 2, mid, h2, w2, F(K.w3), oc, 1, 1, 1, 0,
               F(K.s3), F(K.b3), resid, y, nullptr, 1, ws.wscr, Ho, Wo);
      __bf16* tmp = x; x = y; y = tmp;
      Cin = oc; H = h2; W = w2;
    }
    hipMemcpyAsync(cs[s], x, (size_t)2 * Cin * H * W * sizeof(__bf16),
                   hipMemcpyDeviceToDevice, st);
  }

  // FPN
  const int hw[4]  = {56, 28, 14, 7};
  const int ccs[4] = {256, 512, 1024, 2048};
  for (int i = 0; i < 4; ++i)
    run_conv(st, cs[i], 2, ccs[i], hw[i], hw[i], F(B.lw[i]), 256, 1, 1, 1, 0,
             nullptr, F(B.lb[i]), nullptr, lat[i], nullptr, 0, ws.wscr, Ho, Wo);
  for (int i = 2; i >= 0; --i) {
    int total = 2 * 256 * hw[i] * hw[i];
    upadd_kernel<<<(total + 255) / 256, 256, 0, st>>>(lat[i], lat[i + 1], 2, 256, hw[i], hw[i]);
  }
  for (int i = 0; i < 4; ++i)
    run_conv(st, lat[i], 2, 256, hw[i], hw[i], F(B.ow[i]), 256, 3, 3, 1, 1,
             nullptr, F(B.ob[i]), nullptr, feat[i], nullptr, 0, ws.wscr, Ho, Wo);
}

extern "C" void kernel_launch(void* const* d_in, const int* in_sizes, int n_in,
                              void* d_out, int out_size, void* d_ws, size_t ws_size,
                              hipStream_t stream) {
  (void)in_sizes; (void)n_in; (void)out_size; (void)ws_size;

  // ---- bump allocator over d_ws (all bf16 scratch) ----
  char* wsb = (char*)d_ws;
  size_t off = 0;
  auto alloc_bf = [&](size_t elems) -> __bf16* {
    __bf16* p = (__bf16*)(wsb + off);
    off = (off + elems * sizeof(__bf16) + 255) & ~(size_t)255;
    return p;
  };

  WSBuf ws;
  ws.wscr = alloc_bf(2400000);   // largest bf16 weight: 512*4608 = 2,359,296 (Kpad%64==0)
  ws.x    = alloc_bf(1605632);
  ws.y    = alloc_bf(1605632);
  ws.t1   = alloc_bf(802816);
  ws.t2   = alloc_bf(401408);
  ws.r    = alloc_bf(1605632);
  ws.cat  = alloc_bf(3211264);
  __bf16* in_rgb = alloc_bf(301056);
  __bf16* in_dep = alloc_bf(301056);

  const size_t cs_sz[4]  = {1605632, 802816, 401408, 200704};
  const size_t lat_sz[4] = {1605632, 401408, 100352, 25088};
  __bf16 *csR[4], *csD[4], *latR[4], *latD[4], *ftR[4], *ftD[4];
  for (int i = 0; i < 4; ++i) { csR[i]  = alloc_bf(cs_sz[i]);  csD[i]  = alloc_bf(cs_sz[i]); }
  for (int i = 0; i < 4; ++i) { latR[i] = alloc_bf(lat_sz[i]); latD[i] = alloc_bf(lat_sz[i]); }
  for (int i = 0; i < 4; ++i) { ftR[i]  = alloc_bf(lat_sz[i]); ftD[i]  = alloc_bf(lat_sz[i]); }

  // ---- split stacked input into bf16 rgb / depth ----
  { int total = 2 * 3 * 224 * 224;
    split6_kernel<<<(total + 255) / 256, 256, 0, stream>>>(
        (const float*)d_in[0], in_rgb, in_dep, 2, 224 * 224); }

  // ---- param index maps (jax pytree flatten order) ----
  BBIdx R, D;
  fill_bb(R, 1);          // params_rgb: leaves 1..175
  fill_bb(D, 1 + 175);    // params_depth: leaves 176..350
  const int fw_idx = 1 + 2 * 175;      // 351
  const int fb_idx = fw_idx + 1;       // 352

  run_backbone(stream, R, d_in, in_rgb, ws, csR, latR, ftR);
  run_backbone(stream, D, d_in, in_dep, ws, csD, latD, ftD);

  // ---- fusion head: 1x1 conv over channel-concat, fp32 outputs ----
  const float* fw = (const float*)d_in[fw_idx];
  const float* fb = (const float*)d_in[fb_idx];
  const int hw4[4] = {56, 28, 14, 7};
  const size_t oo[5] = {0, 1605632, 2007040, 2107392, 2132480};
  float* out = (float*)d_out;
  int Ho, Wo;
  for (int i = 0; i < 4; ++i) {
    int total = 2 * 512 * hw4[i] * hw4[i];
    catcopy_kernel<<<(total + 255) / 256, 256, 0, stream>>>(
        ftR[i], ftD[i], ws.cat, 2, 256, hw4[i] * hw4[i]);
    run_conv(stream, ws.cat, 2, 512, hw4[i], hw4[i], fw, 256, 1, 1, 1, 0,
             nullptr, fb, nullptr, nullptr, out + oo[i], 0, ws.wscr, Ho, Wo);
  }
  // level 4 = level 3 strided (1x1 conv commutes with ::2 subsampling)
  { int total = 2 * 256 * 4 * 4;
    sub2_kernel<<<(total + 255) / 256, 256, 0, stream>>>(
        out + oo[3], out + oo[4], 2, 256, 7, 7, 4, 4); }
}
